// CorItN_SE_Single_44676249813414
// MI455X (gfx1250) — matrix operand order repaired and checked by reading, not compiled, not run on hardware
//
#include <hip/hip_runtime.h>
#include <math.h>

// ---------------------------------------------------------------------------
// IterNorm whitening for MI455X (gfx1250, wave32, WMMA + TDM).
//   X: [64, 256, 56, 56] f32.  M = 64*3136 = 200704 samples per channel.
//   Heavy GEMMs (cov 256x256xK=200704, apply 256x200704xK=256) run on
//   v_wmma_f32_16x16x32_bf16 with a hi/lo bf16 split (3 WMMAs ~ fp32 product).
//   Covariance staging uses the Tensor Data Mover (tensor_load_to_lds),
//   double-buffered, gated by s_wait_tensorcnt.
//   Newton-Schulz (tiny 256x256 GEMMs) runs on v_wmma_f32_16x16x4_f32.
// ---------------------------------------------------------------------------

typedef __attribute__((ext_vector_type(16))) __bf16 v16bf;
typedef __attribute__((ext_vector_type(8)))  __bf16 v8bf;
typedef __attribute__((ext_vector_type(4)))  __bf16 v4bf;
typedef __attribute__((ext_vector_type(8)))  float  v8f;
typedef __attribute__((ext_vector_type(2)))  float  v2f;

#define CCH   256
#define NB    64
#define HWSZ  3136
#define MTOT  200704          // NB * HWSZ
#define EPS_SIG 0.001f
#define EPS_BN  1e-5f

union F16x { v16bf v; v8bf h[2]; };

__device__ __forceinline__ v8f wmma_bf16(v16bf a, v16bf b, v8f c) {
    return __builtin_amdgcn_wmma_f32_16x16x32_bf16(false, a, false, b,
                                                   (short)0, c, false, false);
}

// split f32 -> bf16 hi + bf16 lo (x ~= hi + lo), applied after normalization
__device__ __forceinline__ void stash4(float4 v, float mu, float rs,
                                       __bf16* hp, __bf16* lp) {
    float f0 = (v.x - mu) * rs, f1 = (v.y - mu) * rs;
    float f2 = (v.z - mu) * rs, f3 = (v.w - mu) * rs;
    v4bf h, l;
    __bf16 h0 = (__bf16)f0; h[0] = h0; l[0] = (__bf16)(f0 - (float)h0);
    __bf16 h1 = (__bf16)f1; h[1] = h1; l[1] = (__bf16)(f1 - (float)h1);
    __bf16 h2 = (__bf16)f2; h[2] = h2; l[2] = (__bf16)(f2 - (float)h2);
    __bf16 h3 = (__bf16)f3; h[3] = h3; l[3] = (__bf16)(f3 - (float)h3);
    *(v4bf*)hp = h;
    *(v4bf*)lp = l;
}

// ---------------------------------------------------------------------------
// Tensor Data Mover: 2D f32 tile (tile_w elems x tile_h rows) -> LDS.
// D# layout per CDNA5 ISA ch.8: group0 = {flags, lds_addr, global_addr, type},
// group1 = {data_size, tensor dims, tile dims, strides}.
// ---------------------------------------------------------------------------
#if __has_builtin(__builtin_amdgcn_tensor_load_to_lds)
#define HAVE_TDM 1
typedef unsigned int u32x4 __attribute__((ext_vector_type(4)));
typedef int i32x4 __attribute__((ext_vector_type(4)));
typedef int i32x8 __attribute__((ext_vector_type(8)));

__device__ __forceinline__ void tdm_load_tile_f32(unsigned lds_byte_off,
                                                  const float* gptr,
                                                  int tile_w, int tile_h,
                                                  int row_stride_elems) {
    unsigned long long ga = (unsigned long long)(const void*)gptr;
    u32x4 g0;
    g0[0] = 1u;                                    // count=1, user descriptor
    g0[1] = lds_byte_off;                          // D#.lds_addr (bytes)
    g0[2] = (unsigned)ga;                          // global_addr[31:0]
    g0[3] = (unsigned)((ga >> 32) & 0x01ffffffu)   // global_addr[56:32]
            | (2u << 30);                          // type = 2 ("image")
    i32x8 g1;
    unsigned sr = (unsigned)row_stride_elems;
    g1[0] = (int)(2u << 16);                       // data_size = 2 -> 4 bytes
    g1[1] = (int)((sr & 0xffffu) << 16);           // tensor_dim0[15:0]
    g1[2] = (int)(((sr >> 16) & 0xffffu)           // tensor_dim0[31:16]
            | (0x4000u << 16));                    // tensor_dim1[15:0] = 16384
    g1[3] = (int)(((unsigned)tile_w & 0xffffu) << 16); // tile_dim0 (dim1 hi = 0)
    g1[4] = (int)((unsigned)tile_h & 0xffffu);     // tile_dim1 (tile_dim2 = 0)
    g1[5] = (int)sr;                               // tensor_dim0_stride[31:0]
    g1[6] = 0;                                     // stride0[47:32], stride1[15:0]
    g1[7] = 0;                                     // tensor_dim1_stride[47:16]
    i32x4 zz = {0, 0, 0, 0};
#if defined(__clang_major__) && (__clang_major__ >= 23)
    i32x8 z8 = {0, 0, 0, 0, 0, 0, 0, 0};
    __builtin_amdgcn_tensor_load_to_lds(g0, g1, zz, zz, z8, 0);
#else
    __builtin_amdgcn_tensor_load_to_lds(g0, g1, zz, zz, 0);
#endif
}
#endif

// ---------------------------------------------------------------------------
// 1) Per-channel mean / 1/(std+eps). One workgroup per channel.
// ---------------------------------------------------------------------------
__global__ void stats_kernel(const float* __restrict__ X,
                             float* __restrict__ mean,
                             float* __restrict__ rstd) {
    __shared__ float s1s[256], s2s[256];
    const int c = blockIdx.x, t = threadIdx.x;
    float s1 = 0.f, s2 = 0.f;
    for (int n = 0; n < NB; ++n) {
        const float* base = X + ((long)n * CCH + c) * HWSZ;
        for (int i = 4 * t; i < HWSZ; i += 1024) {
            float4 v = *(const float4*)(base + i);
            s1 += v.x + v.y + v.z + v.w;
            s2 += v.x * v.x + v.y * v.y + v.z * v.z + v.w * v.w;
        }
    }
    s1s[t] = s1; s2s[t] = s2;
    __syncthreads();
    for (int off = 128; off > 0; off >>= 1) {
        if (t < off) { s1s[t] += s1s[t + off]; s2s[t] += s2s[t + off]; }
        __syncthreads();
    }
    if (t == 0) {
        float m   = s1s[0] / (float)MTOT;
        float var = (s2s[0] - s1s[0] * m) / (float)(MTOT - 1);
        var = var > 0.f ? var : 0.f;
        mean[c] = m;
        rstd[c] = 1.0f / (sqrtf(var) + EPS_BN);
    }
}

// ---------------------------------------------------------------------------
// 2) Covariance partials: grid = 4 tile positions x 32 K-slices.
//    Each WG: 128x128 tile of xcs@xcs^T over a K-slice of 2 images,
//    TDM double-buffered f32 staging -> bf16 hi/lo -> 24 WMMAs per chunk/wave.
// ---------------------------------------------------------------------------
__global__ void __launch_bounds__(256, 1)
cov_kernel(const float* __restrict__ X,
           const float* __restrict__ mean,
           const float* __restrict__ rstd,
           float* __restrict__ part) {
    __shared__ __attribute__((aligned(64))) __bf16 Ah[128][32];
    __shared__ __attribute__((aligned(64))) __bf16 Al[128][32];
    __shared__ __attribute__((aligned(64))) __bf16 Bh[128][32];
    __shared__ __attribute__((aligned(64))) __bf16 Bl[128][32];
#ifdef HAVE_TDM
    __shared__ __attribute__((aligned(64))) float Af32[2][128][32];  // 2x16KB
    __shared__ __attribute__((aligned(64))) float Bf32[2][128][32];  // 2x16KB
#endif

    const int bx    = blockIdx.x;
    const int pos   = bx & 3;          // 2x2 tile positions over 256x256
    const int slice = bx >> 2;         // 32 K-slices (2 images each)
    const int i0 = (pos >> 1) * 128, j0 = (pos & 1) * 128;
    const int t = threadIdx.x, w = t >> 5, lane = t & 31;
    const int wi = w & 3, wj = w >> 2; // wave tile: 32 rows x 64 cols
    const int r = t >> 1, half = t & 1;
    const int ca = i0 + r, cb = j0 + r;
    const float ma = mean[ca], ra = rstd[ca];
    const float mb = mean[cb], rb = rstd[cb];
    const int fr = lane & 15;
    const int abase = (lane < 16) ? 0 : 16;
    const int kb    = (lane < 16) ? 0 : 16;

    v8f zero = {0.f, 0.f, 0.f, 0.f, 0.f, 0.f, 0.f, 0.f};
    v8f acc[2][4];
#pragma unroll
    for (int a = 0; a < 2; ++a)
#pragma unroll
        for (int b = 0; b < 4; ++b) acc[a][b] = zero;

#ifdef HAVE_TDM
    unsigned ldsA[2], ldsB[2];
    ldsA[0] = (unsigned)(unsigned long long)(void*)&Af32[0][0][0];
    ldsA[1] = (unsigned)(unsigned long long)(void*)&Af32[1][0][0];
    ldsB[0] = (unsigned)(unsigned long long)(void*)&Bf32[0][0][0];
    ldsB[1] = (unsigned)(unsigned long long)(void*)&Bf32[1][0][0];
    if (t < 32) {   // wave 0 issues chunk 0 (EXEC-independent DMA op)
        const int n0 = 2 * slice;
        tdm_load_tile_f32(ldsA[0], X + ((long)n0 * CCH + i0) * HWSZ, 32, 128, HWSZ);
        tdm_load_tile_f32(ldsB[0], X + ((long)n0 * CCH + j0) * HWSZ, 32, 128, HWSZ);
    }
#endif

    for (int cc = 0; cc < 196; ++cc) {
        const int n = 2 * slice + (cc >= 98 ? 1 : 0);
        const int p = (cc % 98) * 32;
#ifdef HAVE_TDM
        const int buf = cc & 1;
        if (t < 32) __builtin_amdgcn_s_wait_tensorcnt(0);  // chunk cc in LDS
        __syncthreads();
        if (t < 32 && cc + 1 < 196) {    // prefetch chunk cc+1 via TDM
            const int n2 = 2 * slice + ((cc + 1) >= 98 ? 1 : 0);
            const int p2 = ((cc + 1) % 98) * 32;
            tdm_load_tile_f32(ldsA[buf ^ 1], X + ((long)n2 * CCH + i0) * HWSZ + p2,
                              32, 128, HWSZ);
            tdm_load_tile_f32(ldsB[buf ^ 1], X + ((long)n2 * CCH + j0) * HWSZ + p2,
                              32, 128, HWSZ);
        }
        {   // normalize + bf16 hi/lo split from staged f32 tiles
            const float4* pa = (const float4*)&Af32[buf][r][half * 16];
            const float4* pb = (const float4*)&Bf32[buf][r][half * 16];
#pragma unroll
            for (int q = 0; q < 4; ++q) {
                stash4(pa[q], ma, ra, &Ah[r][half * 16 + q * 4], &Al[r][half * 16 + q * 4]);
                stash4(pb[q], mb, rb, &Bh[r][half * 16 + q * 4], &Bl[r][half * 16 + q * 4]);
            }
        }
#else
        __syncthreads();
        {   // fallback: vector-load staging straight from global
            const float4* pa = (const float4*)(X + (((long)n * CCH + ca) * HWSZ + p + half * 16));
            const float4* pb = (const float4*)(X + (((long)n * CCH + cb) * HWSZ + p + half * 16));
#pragma unroll
            for (int q = 0; q < 4; ++q) {
                stash4(pa[q], ma, ra, &Ah[r][half * 16 + q * 4], &Al[r][half * 16 + q * 4]);
                stash4(pb[q], mb, rb, &Bh[r][half * 16 + q * 4], &Bl[r][half * 16 + q * 4]);
            }
        }
        if (cc + 1 < 196) {   // global_prefetch_b8 for next chunk
            const int n2 = 2 * slice + ((cc + 1) >= 98 ? 1 : 0);
            const int p2 = ((cc + 1) % 98) * 32;
            __builtin_prefetch(X + (((long)n2 * CCH + ca) * HWSZ + p2 + half * 16), 0, 3);
            __builtin_prefetch(X + (((long)n2 * CCH + cb) * HWSZ + p2 + half * 16), 0, 3);
        }
#endif
        __syncthreads();

        // A fragments (contiguous K order matching B: lane<16 -> K 0..15)
        F16x ah[2], al[2];
#pragma unroll
        for (int tr = 0; tr < 2; ++tr) {
            const int rr = wi * 32 + tr * 16 + fr;
            ah[tr].h[0] = *(const v8bf*)&Ah[rr][abase];
            ah[tr].h[1] = *(const v8bf*)&Ah[rr][abase + 8];
            al[tr].h[0] = *(const v8bf*)&Al[rr][abase];
            al[tr].h[1] = *(const v8bf*)&Al[rr][abase + 8];
        }
#pragma unroll
        for (int tc = 0; tc < 4; ++tc) {
            const int col = wj * 64 + tc * 16 + fr;
            v16bf bh = *(const v16bf*)&Bh[col][kb];
            v16bf bl = *(const v16bf*)&Bl[col][kb];
#pragma unroll
            for (int tr = 0; tr < 2; ++tr) {
                acc[tr][tc] = wmma_bf16(ah[tr].v, bh, acc[tr][tc]);   // hi*hi
                acc[tr][tc] = wmma_bf16(ah[tr].v, bl, acc[tr][tc]);   // hi*lo
                acc[tr][tc] = wmma_bf16(al[tr].v, bh, acc[tr][tc]);   // lo*hi
            }
        }
    }

    // write 128x128 partial tile (deterministic slot, no atomics)
    float* ptile = part + (size_t)bx * 16384;
#pragma unroll
    for (int tr = 0; tr < 2; ++tr)
#pragma unroll
        for (int tc = 0; tc < 4; ++tc)
#pragma unroll
            for (int v = 0; v < 8; ++v) {
                const int row = wi * 32 + tr * 16 + ((lane < 16) ? v : v + 8);
                const int col = wj * 64 + tc * 16 + fr;
                ptile[row * 128 + col] = acc[tr][tc][v];
            }
}

// ---------------------------------------------------------------------------
// 3) Reduce partials (fixed order) -> sigma = eps*I + sum/M
// ---------------------------------------------------------------------------
__global__ void cov_reduce_kernel(const float* __restrict__ part,
                                  float* __restrict__ sigma) {
    const int idx = blockIdx.x * 256 + threadIdx.x;
    const int i = idx >> 8, j = idx & 255;
    const int pos = ((i >> 7) << 1) | (j >> 7);
    const int il = i & 127, jl = j & 127;
    float s = 0.f;
    for (int sl = 0; sl < 32; ++sl)
        s += part[(size_t)(sl * 4 + pos) * 16384 + il * 128 + jl];
    float v = s / (float)MTOT;
    if (i == j) v += EPS_SIG;
    sigma[idx] = v;
}

// ---------------------------------------------------------------------------
// 4) trace_inv, sigmaN = sigma*trace_inv, P = I   (single workgroup)
// ---------------------------------------------------------------------------
__global__ void prep_kernel(const float* __restrict__ sigma,
                            float* __restrict__ sigmaN,
                            float* __restrict__ P,
                            float* __restrict__ scal) {
    __shared__ float red[256];
    __shared__ float tinv_s;
    const int t = threadIdx.x;
    red[t] = sigma[t * 256 + t];
    __syncthreads();
    for (int off = 128; off > 0; off >>= 1) {
        if (t < off) red[t] += red[t + off];
        __syncthreads();
    }
    if (t == 0) { tinv_s = 1.0f / red[0]; scal[0] = tinv_s; }
    __syncthreads();
    const float tinv = tinv_s;
    for (int i = t; i < 65536; i += 256) {
        sigmaN[i] = sigma[i] * tinv;
        P[i] = ((i >> 8) == (i & 255)) ? 1.0f : 0.0f;
    }
}

// ---------------------------------------------------------------------------
// 5) 256x256 f32 GEMM:  C = alpha*(A@B) + beta*D   (Newton-Schulz steps)
//    Uses v_wmma_f32_16x16x4_f32; tiny matrices live in L2.
// ---------------------------------------------------------------------------
#if __has_builtin(__builtin_amdgcn_wmma_f32_16x16x4_f32)
#define HAVE_WMMA_F32 1
#endif

__global__ void gemm256_kernel(float* __restrict__ Cm,
                               const float* __restrict__ Am,
                               const float* __restrict__ Bm, float alpha,
                               const float* __restrict__ Dm, float beta) {
#ifdef HAVE_WMMA_F32
    const int bt = blockIdx.x;                   // 16 WGs: 4x4 of 64x64 tiles
    const int r0 = (bt >> 2) * 64, c0 = (bt & 3) * 64;
    const int t = threadIdx.x, w = t >> 5, lane = t & 31;
    const int idx0 = 2 * w;                      // each wave: 2 adjacent tiles
    const int tr = idx0 >> 2, tc0 = idx0 & 3;
    const int fr = lane & 15;
    const int krel = (lane < 16) ? 0 : 2;        // f32 A 16x4 layout
    v8f acc0 = {0.f,0.f,0.f,0.f,0.f,0.f,0.f,0.f};
    v8f acc1 = acc0;
    const int arow  = r0 + tr * 16 + fr;
    const int bcol0 = c0 + tc0 * 16 + fr;
    for (int k = 0; k < 256; k += 4) {
        const int kk = k + krel;
        v2f a;  a.x  = Am[arow * 256 + kk];       a.y  = Am[arow * 256 + kk + 1];
        v2f b0; b0.x = Bm[kk * 256 + bcol0];      b0.y = Bm[(kk + 1) * 256 + bcol0];
        v2f b1; b1.x = Bm[kk * 256 + bcol0 + 16]; b1.y = Bm[(kk + 1) * 256 + bcol0 + 16];
        acc0 = __builtin_amdgcn_wmma_f32_16x16x4_f32(false, a, false, b0, (short)0, acc0, false, false);
        acc1 = __builtin_amdgcn_wmma_f32_16x16x4_f32(false, a, false, b1, (short)0, acc1, false, false);
    }
#pragma unroll
    for (int v = 0; v < 8; ++v) {
        const int row = r0 + tr * 16 + ((lane < 16) ? v : v + 8);
        const int col = c0 + tc0 * 16 + fr;
        float o0 = alpha * acc0[v];
        float o1 = alpha * acc1[v];
        if (beta != 0.0f) {
            o0 += beta * Dm[row * 256 + col];
            o1 += beta * Dm[row * 256 + col + 16];
        }
        Cm[row * 256 + col]      = o0;
        Cm[row * 256 + col + 16] = o1;
    }
#else
    const int gid = blockIdx.x * 256 + threadIdx.x;
    for (int e = gid; e < 65536; e += 4096) {
        const int row = e >> 8, col = e & 255;
        float s = 0.f;
        for (int k = 0; k < 256; ++k) s += Am[row * 256 + k] * Bm[k * 256 + col];
        float o = alpha * s;
        if (beta != 0.0f) o += beta * Dm[e];
        Cm[e] = o;
    }
#endif
}

// ---------------------------------------------------------------------------
// 6) wm = P * sqrt(trace_inv)  ->  bf16 hi/lo
// ---------------------------------------------------------------------------
__global__ void wm_make_kernel(const float* __restrict__ P,
                               const float* __restrict__ scal,
                               unsigned short* __restrict__ whi_u,
                               unsigned short* __restrict__ wlo_u) {
    const int i = blockIdx.x * 256 + threadIdx.x;
    const float s = sqrtf(scal[0]);
    const float v = P[i] * s;
    __bf16 h = (__bf16)v;
    ((__bf16*)whi_u)[i] = h;
    ((__bf16*)wlo_u)[i] = (__bf16)(v - (float)h);
}

// ---------------------------------------------------------------------------
// 7) Apply: out = wm @ xcs, streaming X with on-the-fly normalize + split.
//    Grid = 3136 WGs (one 64-column block of one image each); each WG does
//    the full 256-row output for its columns (K = 256 in 8 chunks of 32).
// ---------------------------------------------------------------------------
__global__ void __launch_bounds__(256, 1)
apply_kernel(const float* __restrict__ X,
             const float* __restrict__ mean,
             const float* __restrict__ rstd,
             const unsigned short* __restrict__ whi_u,
             const unsigned short* __restrict__ wlo_u,
             float* __restrict__ out) {
    const __bf16* whi = (const __bf16*)whi_u;
    const __bf16* wlo = (const __bf16*)wlo_u;
    __shared__ __attribute__((aligned(64))) __bf16 Bh[64][32];
    __shared__ __attribute__((aligned(64))) __bf16 Bl[64][32];

    const int mb = blockIdx.x;
    const int n = mb / 49, p = (mb % 49) * 64;   // 3136/64 = 49 blocks/image
    const int t = threadIdx.x, w = t >> 5, lane = t & 31;
    const int chan = t >> 3, seg = t & 7;        // staging: 32 chans x 8 segs
    const int fr = lane & 15;
    const int kb = (lane < 16) ? 0 : 16;

    v8f zero = {0.f, 0.f, 0.f, 0.f, 0.f, 0.f, 0.f, 0.f};
    v8f acc[2][4];
#pragma unroll
    for (int a = 0; a < 2; ++a)
#pragma unroll
        for (int b = 0; b < 4; ++b) acc[a][b] = zero;

    for (int kc = 0; kc < 8; ++kc) {
        const int k0 = kc * 32;
        __syncthreads();
        {   // stage B = xcs[k0..k0+31][p..p+63] transposed to [col][K]
            const int cC = k0 + chan;
            const float mu = mean[cC], rs = rstd[cC];
            const float4* px = (const float4*)(X + (((long)n * CCH + cC) * HWSZ + p + seg * 8));
            float4 v0 = px[0], v1 = px[1];
            float f[8] = {v0.x, v0.y, v0.z, v0.w, v1.x, v1.y, v1.z, v1.w};
#pragma unroll
            for (int j = 0; j < 8; ++j) {
                const float x = (f[j] - mu) * rs;
                __bf16 h = (__bf16)x;
                Bh[seg * 8 + j][chan] = h;
                Bl[seg * 8 + j][chan] = (__bf16)(x - (float)h);
            }
        }
        if (kc + 1 < 8) {   // global_prefetch_b8 for next K chunk
            __builtin_prefetch(X + (((long)n * CCH + k0 + 32 + chan) * HWSZ + p + seg * 8), 0, 3);
        }
        __syncthreads();

        // A fragments straight from global wm (L2-resident, 16B loads)
        const int abase = k0 + ((lane < 16) ? 0 : 16);
        F16x ah[2], al[2];
#pragma unroll
        for (int tr = 0; tr < 2; ++tr) {
            const int rr = w * 32 + tr * 16 + fr;
            const __bf16* ph = whi + rr * 256 + abase;
            const __bf16* pl = wlo + rr * 256 + abase;
            ah[tr].h[0] = *(const v8bf*)ph;
            ah[tr].h[1] = *(const v8bf*)(ph + 8);
            al[tr].h[0] = *(const v8bf*)pl;
            al[tr].h[1] = *(const v8bf*)(pl + 8);
        }
#pragma unroll
        for (int tc = 0; tc < 4; ++tc) {
            v16bf bh = *(const v16bf*)&Bh[tc * 16 + fr][kb];
            v16bf bl = *(const v16bf*)&Bl[tc * 16 + fr][kb];
#pragma unroll
            for (int tr = 0; tr < 2; ++tr) {
                acc[tr][tc] = wmma_bf16(ah[tr].v, bh, acc[tr][tc]);
                acc[tr][tc] = wmma_bf16(ah[tr].v, bl, acc[tr][tc]);
                acc[tr][tc] = wmma_bf16(al[tr].v, bh, acc[tr][tc]);
            }
        }
    }

#pragma unroll
    for (int tr = 0; tr < 2; ++tr)
#pragma unroll
        for (int tc = 0; tc < 4; ++tc)
#pragma unroll
            for (int v = 0; v < 8; ++v) {
                const int row = w * 32 + tr * 16 + ((lane < 16) ? v : v + 8);
                const int col = tc * 16 + fr;
                out[((long)n * CCH + row) * HWSZ + p + col] = acc[tr][tc][v];
            }
}

// ---------------------------------------------------------------------------
// Host launcher. Workspace layout (floats):
//   [0,256) mean | [256,512) rstd | [512] trace_inv | 1024+ : sigma, sigmaN,
//   P, P2, P3, Pn (6 x 65536 f32), wm hi/lo (2 x 65536 bf16),
//   cov partials 4*32*16384 f32.  Total ~9.8 MB.
// ---------------------------------------------------------------------------
extern "C" void kernel_launch(void* const* d_in, const int* in_sizes, int n_in,
                              void* d_out, int out_size, void* d_ws, size_t ws_size,
                              hipStream_t stream) {
    (void)in_sizes; (void)n_in; (void)out_size; (void)ws_size;
    const float* X = (const float*)d_in[0];
    float* out = (float*)d_out;

    float* W      = (float*)d_ws;
    float* mean   = W;
    float* rstd   = W + 256;
    float* scal   = W + 512;
    float* sigma  = W + 1024;
    float* sigmaN = sigma  + 65536;
    float* P      = sigmaN + 65536;
    float* P2     = P      + 65536;
    float* P3     = P2     + 65536;
    float* Pn     = P3     + 65536;
    unsigned short* whi = (unsigned short*)(Pn + 65536);
    unsigned short* wlo = whi + 65536;
    float* part   = (float*)(wlo + 65536);

    stats_kernel<<<256, 256, 0, stream>>>(X, mean, rstd);
    cov_kernel<<<128, 256, 0, stream>>>(X, mean, rstd, part);
    cov_reduce_kernel<<<256, 256, 0, stream>>>(part, sigma);
    prep_kernel<<<1, 256, 0, stream>>>(sigma, sigmaN, P, scal);

    float* Pa = P;
    float* Pb = Pn;
    for (int it = 0; it < 5; ++it) {
        gemm256_kernel<<<16, 256, 0, stream>>>(P2, Pa, Pa, 1.0f, Pa, 0.0f);
        gemm256_kernel<<<16, 256, 0, stream>>>(P3, P2, Pa, 1.0f, Pa, 0.0f);
        gemm256_kernel<<<16, 256, 0, stream>>>(Pb, P3, sigmaN, -0.5f, Pa, 1.5f);
        float* tmp = Pa; Pa = Pb; Pb = tmp;
    }

    wm_make_kernel<<<256, 256, 0, stream>>>(Pa, scal, whi, wlo);
    apply_kernel<<<3136, 256, 0, stream>>>(X, mean, rstd, whi, wlo, out);
}
